// Attend_1726576856263
// MI455X (gfx1250) — compile-verified, matrix operand order checked by reading, not run on hardware
//
#include <hip/hip_runtime.h>
#include <cfloat>
#include <cmath>
#include <cstdint>

typedef _Float16 h16;
typedef __attribute__((ext_vector_type(16))) _Float16 v16h;
typedef __attribute__((ext_vector_type(8)))  float    v8f;

#define NSEQ     1024
#define DHEAD    64
#define BQ       128      // query rows per workgroup
#define WQ       16       // query rows per wave
#define BJ       32       // keys per j-iteration (K=32 for the PV WMMA)
#define NWAVES   8
#define NTHREADS 256

// A/B fragment dword r (0..7) + lane-half h (0/1) -> first K index of the
// packed f16 pair, per the CDNA5 16-bit A-matrix 16x32 VGPR layout.
__device__ __forceinline__ int kmap(int r, int h) {
    return 2 * r + ((r >= 4) ? 8 : 0) + 8 * h;
}

__device__ __forceinline__ v8f fz8() {
    v8f z;
#pragma unroll
    for (int i = 0; i < 8; ++i) z[i] = 0.0f;
    return z;
}

// Generic pointer to an LDS object: low 32 bits are the LDS offset.
__device__ __forceinline__ uint32_t lds_addr(const void* p) {
    return (uint32_t)(uintptr_t)p;
}

// CDNA5 async global->LDS DMA (GVS mode: SADDR base + 32-bit VGPR byte offset).
__device__ __forceinline__ void async_copy_b128(uint32_t lds_dst, uint32_t byte_off,
                                                const float* sbase) {
    asm volatile("global_load_async_to_lds_b128 %0, %1, %2"
                 :: "v"(lds_dst), "v"(byte_off), "s"(sbase) : "memory");
}
__device__ __forceinline__ void async_copy_b32(uint32_t lds_dst, uint32_t byte_off,
                                               const float* sbase) {
    asm volatile("global_load_async_to_lds_b32 %0, %1, %2"
                 :: "v"(lds_dst), "v"(byte_off), "s"(sbase) : "memory");
}
__device__ __forceinline__ void wait_async0() {
    asm volatile("s_wait_asynccnt 0x0" ::: "memory");
}

// Max-reduce across the 16 lanes holding one matrix row, using DPP row
// rotations (VALU only -- keeps the LDS pipe free for fragment loads).
__device__ __forceinline__ float row_max16(float x) {
    int xi;
    xi = __builtin_amdgcn_update_dpp(0, __builtin_bit_cast(int, x), 0x121, 0xF, 0xF, true);
    x  = fmaxf(x, __builtin_bit_cast(float, xi));
    xi = __builtin_amdgcn_update_dpp(0, __builtin_bit_cast(int, x), 0x122, 0xF, 0xF, true);
    x  = fmaxf(x, __builtin_bit_cast(float, xi));
    xi = __builtin_amdgcn_update_dpp(0, __builtin_bit_cast(int, x), 0x124, 0xF, 0xF, true);
    x  = fmaxf(x, __builtin_bit_cast(float, xi));
    xi = __builtin_amdgcn_update_dpp(0, __builtin_bit_cast(int, x), 0x128, 0xF, 0xF, true);
    x  = fmaxf(x, __builtin_bit_cast(float, xi));
    return x;
}

__global__ __launch_bounds__(NTHREADS)
void attn_fwd_wmma(const float* __restrict__ q, const float* __restrict__ k,
                   const float* __restrict__ v, const float* __restrict__ bias,
                   float* __restrict__ out) {
    __shared__ float kf_lds[2][BJ * DHEAD];      // keys, row-major (j, d), dbl buf
    __shared__ float vT_lds[2][DHEAD * BJ];      // values transposed (d, j), dbl buf
    __shared__ h16   p_lds[NWAVES * WQ * BJ];    // per-wave P transpose scratch

    const int tid  = threadIdx.x;
    const int wave = tid >> 5;
    const int lane = tid & 31;
    const int n16  = lane & 15;    // column index inside a 16-wide tile
    const int h    = lane >> 4;    // lane half (selects K subset / row offset)

    const int iblk  = blockIdx.x;              // 0..7
    const int bh    = blockIdx.y;              // 0..63 (b*H + h)
    const int wg_i0 = iblk * BQ;
    const int i0    = wg_i0 + wave * WQ;       // this wave's first query row

    const float scale = 0.125f;                // 1/sqrt(64)

    const size_t base_qk = (size_t)bh * NSEQ * DHEAD;
    const size_t base_b  = (size_t)bh * NSEQ * NSEQ;

    // ---- Q fragments (A layout), fp32 -> f16 on the fly -------------------
    v16h qa[2];
    {
        const float* qrow = q + base_qk + (size_t)(i0 + n16) * DHEAD;
#pragma unroll
        for (int f = 0; f < 2; ++f) {
#pragma unroll
            for (int r = 0; r < 8; ++r) {
                const int d0 = f * 32 + kmap(r, h);
                qa[f][2 * r]     = (h16)qrow[d0];
                qa[f][2 * r + 1] = (h16)qrow[d0 + 1];
            }
        }
    }

    // all-ones B fragment: row-sum of P via WMMA (P x ones -> l in C layout)
    v16h ones;
#pragma unroll
    for (int i = 0; i < 16; ++i) ones[i] = (h16)1.0f;

    // ---- accumulators & online-softmax state ------------------------------
    v8f o[4];
#pragma unroll
    for (int t = 0; t < 4; ++t) o[t] = fz8();
    v8f lacc = fz8();                          // softmax denominator, C layout
    float mrow[8];
#pragma unroll
    for (int r = 0; r < 8; ++r) mrow[r] = -FLT_MAX;

    const int njb = (wg_i0 + BQ) / BJ;         // uniform across the workgroup

    // ---- async stage of one 32-key K/V tile into LDS buffer `buf` ---------
    auto stage = [&](int j0, int buf) {
        const float* kb_ = k + base_qk + (size_t)j0 * DHEAD;
        const float* vb_ = v + base_qk + (size_t)j0 * DHEAD;
        // K row-major: 2048 dwords, b128 per lane -> 2 DMA ops per thread
#pragma unroll
        for (int i = 0; i < (BJ * DHEAD) / (NTHREADS * 4); ++i) {
            const int e = tid * 4 + i * NTHREADS * 4;
            async_copy_b128(lds_addr(&kf_lds[buf][e]), (uint32_t)(e * 4), kb_);
        }
        // V transposed in the DMA: per-lane LDS dest = (d, j); global contiguous
#pragma unroll
        for (int i = 0; i < (BJ * DHEAD) / NTHREADS; ++i) {
            const int e  = tid + i * NTHREADS;
            const int jr = e / DHEAD;
            const int c  = e % DHEAD;
            async_copy_b32(lds_addr(&vT_lds[buf][c * BJ + jr]), (uint32_t)(e * 4), vb_);
        }
    };

    stage(0, 0);
    int cur = 0;

    for (int jb = 0; jb < njb; ++jb) {
        const int j0 = jb * BJ;

        wait_async0();          // this wave's DMA for tile jb is in LDS
        __syncthreads();        // everyone's DMA visible; prev buffer free

        if (jb + 1 < njb) stage((jb + 1) * BJ, cur ^ 1);   // overlap next tile

        if (j0 <= i0 + WQ - 1) {               // wave-uniform causal skip
            // prefetch next iteration's 16x32 bias tile (one line per row)
            if (jb + 1 < njb) {
                const float* bp = bias + base_b + (size_t)(i0 + n16) * NSEQ
                                  + (size_t)(j0 + BJ) + h * 16;
                __builtin_prefetch(bp, 0, 0);
            }

            const float* kbuf = kf_lds[cur];
            const float* vbuf = vT_lds[cur];

            // ---- S = Q K^T (+scale +bias +causal), two 16-key subtiles ----
            v8f svals[2];
#pragma unroll
            for (int jt = 0; jt < 2; ++jt) {
                const float* krow = &kbuf[(jt * 16 + n16) * DHEAD];
                v8f s = fz8();
#pragma unroll
                for (int f = 0; f < 2; ++f) {
                    v16h kb;
#pragma unroll
                    for (int r = 0; r < 8; ++r) {
                        const int d0 = f * 32 + kmap(r, h);
                        kb[2 * r]     = (h16)krow[d0];
                        kb[2 * r + 1] = (h16)krow[d0 + 1];
                    }
                    s = __builtin_amdgcn_wmma_f32_16x16x32_f16(
                            false, qa[f], false, kb, (short)0, s, false, false);
                }
#pragma unroll
                for (int r = 0; r < 8; ++r) {
                    const int row = i0 + r + 8 * h;
                    const int col = j0 + jt * 16 + n16;
                    // bias is stream-once: non-temporal keeps K/V resident in L2
                    float val = s[r] * scale + __builtin_nontemporal_load(
                                    bias + base_b + (size_t)row * NSEQ + col);
                    if (col > row) val = -FLT_MAX;   // causal: strictly above diag
                    s[r] = val;
                }
                svals[jt] = s;
            }

            // ---- online softmax: DPP row-max, WMMA row-sum ----------------
            float alpha[8];
#pragma unroll
            for (int r = 0; r < 8; ++r) {
                const float t    = row_max16(fmaxf(svals[0][r], svals[1][r]));
                const float mnew = fmaxf(mrow[r], t);
                alpha[r] = __expf(mrow[r] - mnew);
                mrow[r]  = mnew;
            }

            h16* pw = &p_lds[wave * WQ * BJ];
#pragma unroll
            for (int r = 0; r < 8; ++r) {
                const float p0 = __expf(svals[0][r] - mrow[r]);
                const float p1 = __expf(svals[1][r] - mrow[r]);
                const int   M  = r + 8 * h;
                pw[M * BJ + n16]      = (h16)p0;   // C-layout -> (M, j) scratch
                pw[M * BJ + 16 + n16] = (h16)p1;
            }

            // ---- rescale O and l, then O += P V, l += P * ones ------------
#pragma unroll
            for (int t = 0; t < 4; ++t)
#pragma unroll
                for (int r = 0; r < 8; ++r) o[t][r] *= alpha[r];
#pragma unroll
            for (int r = 0; r < 8; ++r) lacc[r] *= alpha[r];

            v16h pa;
            {
                const h16* pr = &p_lds[wave * WQ * BJ + n16 * BJ];
#pragma unroll
                for (int r = 0; r < 8; ++r) {
                    const int jj = kmap(r, h);
                    pa[2 * r]     = pr[jj];
                    pa[2 * r + 1] = pr[jj + 1];
                }
            }
            lacc = __builtin_amdgcn_wmma_f32_16x16x32_f16(
                       false, pa, false, ones, (short)0, lacc, false, false);
#pragma unroll
            for (int t = 0; t < 4; ++t) {
                v16h vb;
                const float* vcol = &vbuf[(t * 16 + n16) * BJ];
#pragma unroll
                for (int r = 0; r < 8; ++r) {
                    const int jj = kmap(r, h);
                    vb[2 * r]     = (h16)vcol[jj];
                    vb[2 * r + 1] = (h16)vcol[jj + 1];
                }
                o[t] = __builtin_amdgcn_wmma_f32_16x16x32_f16(
                           false, pa, false, vb, (short)0, o[t], false, false);
            }
        }
        cur ^= 1;
    }

    // ---- normalize and store (non-temporal: output is never re-read) ------
#pragma unroll
    for (int r = 0; r < 8; ++r) {
        const float inv = 1.0f / lacc[r];
        const int   row = i0 + r + 8 * h;
        float* orow = out + base_qk + (size_t)row * DHEAD;
#pragma unroll
        for (int t = 0; t < 4; ++t)
            __builtin_nontemporal_store(o[t][r] * inv, orow + t * 16 + n16);
    }
}

extern "C" void kernel_launch(void* const* d_in, const int* in_sizes, int n_in,
                              void* d_out, int out_size, void* d_ws, size_t ws_size,
                              hipStream_t stream) {
    const float* q    = (const float*)d_in[0];
    const float* k    = (const float*)d_in[1];
    const float* v    = (const float*)d_in[2];
    const float* bias = (const float*)d_in[3];
    float* out = (float*)d_out;

    dim3 grid(NSEQ / BQ, 4 * 16);   // (i-blocks, B*H)
    dim3 block(NTHREADS);
    attn_fwd_wmma<<<grid, block, 0, stream>>>(q, k, v, bias, out);
}
